// SafeGRU_18554258718734
// MI455X (gfx1250) — matrix operand order repaired and checked by reading, not compiled, not run on hardware
//
#include <hip/hip_runtime.h>
#include <stdint.h>

// ---------------------------------------------------------------------------
// 2-layer GRU forward on MI455X (gfx1250, wave32, WMMA bf16).
//
//   gru_prep_kernel : fp32 weights -> bf16 (workspace), reset sync flags.
//   gru_pipe_kernel : 16 workgroups, 256 threads (8 waves) each.
//       blocks 0..7  = layer 0 (producers), batch tile mt = blockIdx & 7
//       blocks 8..15 = layer 1 (consumers), same batch tiles
//     Layer-0 blocks stream h0_t (bf16) through L2 staging and publish with
//     agent-scope release; layer-1 blocks acquire-spin and consume.
//
//   Per wave: owns gate n-tiles {w, w+8, w+16} == column slice [16w,16w+16)
//   of r, z, n gates AND of h, so all gate math + z*h recurrence is in-lane.
//   W_hh B-fragments are register-resident (time-invariant, 96 VGPRs);
//   r/z gates chain gi and gh GEMMs into a single WMMA accumulator.
//
// Workspace layout (d_ws), needs >= 34 MB:
//   [0 .. 384KB)     bf16 weights: [layer][{W_ih,W_hh}][384][128]
//   [384KB .. +32B)  int flags[8]  (per-mtile progress counter)
//   [512KB .. +32MB) bf16 staging: [mt][t][16][128]
// ---------------------------------------------------------------------------

#define BATCH 128
#define SEQT  1024
#define INF   128
#define HDIM  128
#define G3    384           // 3*H
#define WPER  49152         // elements per weight matrix (384*128)

#define WS_FLAG_OFF  ((size_t)4 * WPER * 2)   // 393216 bytes
#define WS_STG_OFF   ((size_t)512 * 1024)     // 524288 bytes

typedef __attribute__((ext_vector_type(16))) __bf16 v16bf;
typedef __attribute__((ext_vector_type(8)))  float  v8f;

static __device__ __forceinline__ uint16_t f2bf(float f) {
    union { float f; uint32_t u; } v; v.f = f;
    uint32_t r = v.u + 0x7FFFu + ((v.u >> 16) & 1u);   // round-to-nearest-even
    return (uint16_t)(r >> 16);
}

static __device__ __forceinline__ float sigmoidf_(float x) {
    return 1.0f / (1.0f + __expf(-x));
}

// Load a WMMA bf16 fragment as two 16B reads (ds_load_b128 / global_load_b128).
static __device__ __forceinline__ v16bf ld_frag2(const uint16_t* p0,
                                                 const uint16_t* p1) {
    union { uint4 q[2]; v16bf v; } u;
    u.q[0] = *(const uint4*)p0;
    u.q[1] = *(const uint4*)p1;
    return u.v;
}

#define WMMA_BF16(A, B, C) \
    __builtin_amdgcn_wmma_f32_16x16x32_bf16(false, (A), false, (B), (short)0, \
                                            (C), false, false)

// ---------------------------------------------------------------------------
// Prep: convert W_ih/W_hh fp32 -> bf16 into workspace; zero the sync flags.
// ---------------------------------------------------------------------------
__global__ void gru_prep_kernel(const float* __restrict__ Wih,
                                const float* __restrict__ Whh,
                                uint16_t* __restrict__ wbf,
                                int* __restrict__ flags) {
    int i = blockIdx.x * blockDim.x + threadIdx.x;
    if (i < 4 * WPER) {
        int mat = i / WPER;          // [l][which]
        int r   = i % WPER;
        int l   = mat >> 1;
        const float* src = (mat & 1) ? Whh : Wih;
        wbf[i] = f2bf(src[l * WPER + r]);
    }
    if (i < 8) flags[i] = 0;
}

// ---------------------------------------------------------------------------
// Main pipelined GRU kernel.
// ---------------------------------------------------------------------------
__global__ __launch_bounds__(256, 1)
void gru_pipe_kernel(const float* __restrict__ x,
                     const float* __restrict__ b_ih,
                     const float* __restrict__ b_hh,
                     const uint16_t* __restrict__ wbf,
                     uint16_t* __restrict__ stg,
                     int* __restrict__ flags,
                     float* __restrict__ out,
                     float* __restrict__ hfin) {
    // Dynamic LDS: [W_ih bf16 96KB][abuf 4KB][hbuf 4KB] = 104KB
    extern __shared__ __align__(16) uint16_t smem[];
    uint16_t* wih  = smem;               // 49152 : W_ih bf16 [384][128]
    uint16_t* abuf = smem + WPER;        // 16x128 bf16 A-operand (x_t / h0_t)
    uint16_t* hbuf = abuf + 2048;        // 16x128 bf16 own hidden state

    const int layer = blockIdx.x >> 3;   // 0: producer, 1: consumer
    const int mt    = blockIdx.x & 7;    // batch tile
    const int m0    = mt * 16;

    const int tid  = threadIdx.x;
    const int wv   = tid >> 5;           // wave 0..7
    const int lane = tid & 31;
    const int half = lane >> 4;          // 0: lanes 0-15, 1: lanes 16-31
    const int lrow = lane & 15;
    const int col  = wv * 16 + lrow;     // owned h / gate column (0..127)

    // ---- W_ih (this layer) -> LDS; W_hh fragments -> registers -------------
    {
        const uint4* src = (const uint4*)(wbf + (size_t)layer * 2 * WPER);
        uint4*       dst = (uint4*)wih;  // 6144 uint4
        for (int i = tid; i < 6144; i += 256) dst[i] = src[i];
    }
    v16bf whhf[3][4];                    // 96 VGPRs, time-invariant
    {
        const uint16_t* wsrc = wbf + (size_t)layer * 2 * WPER + WPER;
#pragma unroll
        for (int j = 0; j < 3; ++j) {
            const uint16_t* wp = wsrc + ((wv + 8 * j) * 16 + lrow) * 128
                                      + half * 16;
#pragma unroll
            for (int c = 0; c < 4; ++c)
                whhf[j][c] = ld_frag2(wp + c * 32, wp + c * 32 + 8);
        }
    }
    // zero hidden-state buffer (h0 = 0)
    {
        uint4 z; z.x = z.y = z.z = z.w = 0u;
        ((uint4*)hbuf)[tid] = z;
    }
    __syncthreads();

    // ---- per-lane gate biases (constant over time) -------------------------
    const float* bi = b_ih + layer * G3;
    const float* bh = b_hh + layer * G3;
    const float br   = bi[col]       + bh[col];        // r-gate: gi+gh fused
    const float bz   = bi[128 + col] + bh[128 + col];  // z-gate: gi+gh fused
    const float bi_n = bi[256 + col];
    const float bh_n = bh[256 + col];

    // persistent hidden state, element (m = r + 8*half, col) in h[r]
    float h[8];
#pragma unroll
    for (int r = 0; r < 8; ++r) h[r] = 0.0f;

    for (int t = 0; t < SEQT; ++t) {
        // ==== 1. hidden-state A-fragments (h_{t-1}); overlaps abuf fill =====
        v16bf afh[4];
        {
            const uint16_t* hb = hbuf + lrow * 128 + half * 16;
#pragma unroll
            for (int c = 0; c < 4; ++c)
                afh[c] = ld_frag2(hb + c * 32, hb + c * 32 + 8);
        }

        // ==== 2. stage A-operand of the W_ih GEMM into abuf =================
        if (layer == 0) {
            // x tile [16][128] fp32 -> bf16; each thread handles 8 elements
            const int row = tid >> 4;
            const int c8  = (tid & 15) << 3;
            const float* xp = x + ((size_t)(m0 + row) * SEQT + t) * INF + c8;
            float4 f0 = *(const float4*)xp;
            float4 f1 = *(const float4*)(xp + 4);
            if (t + 1 < SEQT) __builtin_prefetch(xp + INF, 0, 1);
            union { uint16_t hh[8]; uint4 q; } pk;
            pk.hh[0] = f2bf(f0.x); pk.hh[1] = f2bf(f0.y);
            pk.hh[2] = f2bf(f0.z); pk.hh[3] = f2bf(f0.w);
            pk.hh[4] = f2bf(f1.x); pk.hh[5] = f2bf(f1.y);
            pk.hh[6] = f2bf(f1.z); pk.hh[7] = f2bf(f1.w);
            *(uint4*)(abuf + row * 128 + c8) = pk.q;
        } else {
            if (tid == 0) {
                while (__hip_atomic_load(flags + mt, __ATOMIC_ACQUIRE,
                                         __HIP_MEMORY_SCOPE_AGENT) <= t) {
                    __builtin_amdgcn_s_sleep(1);
                }
            }
            __syncthreads();
            const uint4* sp = (const uint4*)stg + ((size_t)mt * SEQT + t) * 256;
            ((uint4*)abuf)[tid] = sp[tid];
        }
        __syncthreads();

        // ==== 3. GEMMs ======================================================
        // r/z gates: acc = h*W_hh^T (regs) then += A*W_ih^T (LDS), fused.
        // n gate   : gh and gi kept separate (r multiplies only the gh part).
        v16bf afx[4];
        {
            const uint16_t* ab = abuf + lrow * 128 + half * 16;
#pragma unroll
            for (int c = 0; c < 4; ++c)
                afx[c] = ld_frag2(ab + c * 32, ab + c * 32 + 8);
        }

        v8f acc_r = {0.f, 0.f, 0.f, 0.f, 0.f, 0.f, 0.f, 0.f};
        v8f acc_z = acc_r, ghn = acc_r, gin = acc_r;
#pragma unroll
        for (int c = 0; c < 4; ++c) ghn   = WMMA_BF16(afh[c], whhf[2][c], ghn);
#pragma unroll
        for (int c = 0; c < 4; ++c) acc_r = WMMA_BF16(afh[c], whhf[0][c], acc_r);
#pragma unroll
        for (int c = 0; c < 4; ++c) acc_z = WMMA_BF16(afh[c], whhf[1][c], acc_z);
#pragma unroll
        for (int j = 0; j < 3; ++j) {
            const uint16_t* wp = wih + ((wv + 8 * j) * 16 + lrow) * 128
                                     + half * 16;
            v8f* accp = (j == 0) ? &acc_r : (j == 1) ? &acc_z : &gin;
            v8f  a    = *accp;
#pragma unroll
            for (int c = 0; c < 4; ++c) {
                v16bf bfr = ld_frag2(wp + c * 32, wp + c * 32 + 8);
                a = WMMA_BF16(afx[c], bfr, a);
            }
            *accp = a;
        }
        __syncthreads();   // abuf/hbuf reads done -> safe to rewrite below

        // ==== 4. gate math (fp32, fully in-lane) ============================
#pragma unroll
        for (int r = 0; r < 8; ++r) {
            float rg = sigmoidf_(acc_r[r] + br);
            float zg = sigmoidf_(acc_z[r] + bz);
            float ng = tanhf(gin[r] + bi_n + rg * (ghn[r] + bh_n));
            h[r] = (1.0f - zg) * ng + zg * h[r];
        }
        {
            uint16_t* hp = hbuf + (half * 8) * 128 + col;
#pragma unroll
            for (int r = 0; r < 8; ++r) hp[r * 128] = f2bf(h[r]);
        }
        __syncthreads();

        // ==== 5. publish / emit =============================================
        if (layer == 0) {
            uint4* sp = (uint4*)stg + ((size_t)mt * SEQT + t) * 256;
            sp[tid] = ((const uint4*)hbuf)[tid];
            __threadfence();
            __syncthreads();
            if (tid == 0)
                __hip_atomic_store(flags + mt, t + 1, __ATOMIC_RELEASE,
                                   __HIP_MEMORY_SCOPE_AGENT);
        } else {
            float* op = out + ((size_t)(m0 + half * 8) * SEQT + t) * HDIM + col;
#pragma unroll
            for (int r = 0; r < 8; ++r)
                op[(size_t)r * SEQT * HDIM] = h[r];
        }
    }

    // ==== h_final[layer, m, col] ===========================================
    {
        float* hf = hfin + ((size_t)layer * BATCH + m0 + half * 8) * HDIM + col;
#pragma unroll
        for (int r = 0; r < 8; ++r) hf[r * HDIM] = h[r];
    }
}

// ---------------------------------------------------------------------------
extern "C" void kernel_launch(void* const* d_in, const int* in_sizes, int n_in,
                              void* d_out, int out_size, void* d_ws,
                              size_t ws_size, hipStream_t stream) {
    (void)in_sizes; (void)n_in; (void)out_size; (void)ws_size;

    const float* x   = (const float*)d_in[0];
    const float* Wih = (const float*)d_in[1];
    const float* Whh = (const float*)d_in[2];
    const float* bih = (const float*)d_in[3];
    const float* bhh = (const float*)d_in[4];

    float* out  = (float*)d_out;
    float* hfin = out + (size_t)BATCH * SEQT * HDIM;

    uint16_t* wbf   = (uint16_t*)d_ws;
    int*      flags = (int*)((char*)d_ws + WS_FLAG_OFF);
    uint16_t* stg   = (uint16_t*)((char*)d_ws + WS_STG_OFF);

    gru_prep_kernel<<<768, 256, 0, stream>>>(Wih, Whh, wbf, flags);

    const size_t shmem = (size_t)(WPER + 2 * 2048) * sizeof(uint16_t); // 104KB
    (void)hipFuncSetAttribute((const void*)gru_pipe_kernel,
                              hipFuncAttributeMaxDynamicSharedMemorySize,
                              (int)shmem);
    gru_pipe_kernel<<<16, 256, shmem, stream>>>(x, bih, bhh, wbf, stg, flags,
                                                out, hfin);
}
